// RelationExtractionModel_46557445489279
// MI455X (gfx1250) — matrix operand order repaired and checked by reading, not compiled, not run on hardware
//
#include <hip/hip_runtime.h>
#include <hip/hip_bf16.h>
#include <math.h>

// Model dims
#define BB 64
#define TT 256
#define FEATP 928   // 912 padded to multiple of 32
#define FTD 512

typedef __attribute__((ext_vector_type(16))) _Float16 v16h_t;
typedef __attribute__((ext_vector_type(8)))  float    v8f_t;
typedef __attribute__((ext_vector_type(4)))  unsigned int v4u_t;
typedef __attribute__((ext_vector_type(8)))  int      v8i_t;
typedef __attribute__((ext_vector_type(4)))  int      v4i_t;

union AFrag { v16h_t v; _Float16 e[16]; };
union CFrag { v8f_t  v; float    e[8];  };

__device__ __forceinline__ float sigf(float x) { return 1.0f / (1.0f + expf(-x)); }

// Issue a 2D TDM load: tile of tileRows rows x 64 bytes (16 dwords) from global
// (row stride rowStrideDW dwords) into LDS at ldsByteOff. Wave-uniform values only.
__device__ __forceinline__ void tdm_load_2d(const void* gaddr, unsigned ldsByteOff,
                                            unsigned rowStrideDW, unsigned tensorRows,
                                            unsigned tileRows) {
  unsigned long long ga = (unsigned long long)(uintptr_t)gaddr;
  v4u_t g0;
  g0[0] = 1u;                                            // count=1, no gather
  g0[1] = ldsByteOff;                                    // lds_addr [63:32]
  g0[2] = (unsigned)(ga & 0xffffffffu);                  // global_addr[31:0]
  g0[3] = (unsigned)((ga >> 32) & 0x01ffffffu) | (2u << 30);  // addr[56:32] | type=2
  unsigned td0 = rowStrideDW;                            // tensor_dim0 (4B units)
  unsigned td1 = tensorRows;                             // tensor_dim1
  v8i_t g1;
  g1[0] = (int)(2u << 16);                               // data_size=4B
  g1[1] = (int)((td0 & 0xffffu) << 16);                  // td0[15:0] @ bit48
  g1[2] = (int)((td0 >> 16) | ((td1 & 0xffffu) << 16));  // td0[31:16] | td1[15:0]
  g1[3] = (int)((td1 >> 16) | (16u << 16));              // td1[31:16] | tile_dim0=16
  g1[4] = (int)(tileRows & 0xffffu);                     // tile_dim1 | tile_dim2=0
  g1[5] = (int)rowStrideDW;                              // tensor_dim0_stride[31:0]
  g1[6] = 0;                                             // stride0[47:32] | stride1[15:0]
  g1[7] = 0;
  v4i_t g2 = {0, 0, 0, 0};
  v4i_t g3 = {0, 0, 0, 0};
#if defined(__clang_major__) && (__clang_major__ >= 23)
  v8i_t g4 = {0, 0, 0, 0, 0, 0, 0, 0};
  __builtin_amdgcn_tensor_load_to_lds(g0, g1, g2, g3, g4, 0);
#else
  __builtin_amdgcn_tensor_load_to_lds(g0, g1, g2, g3, 0);
#endif
}

// ---------------- utility kernels ----------------
__global__ void cvt_f32_f16_kernel(const float* __restrict__ src, _Float16* __restrict__ dst, int n) {
  int i = blockIdx.x * 256 + threadIdx.x;
  if (i < n) dst[i] = (_Float16)src[i];
}

// dst[n*K+k] = src[k*N+n]  (weight transpose f32->f16)
__global__ void cvt_transpose_kernel(const float* __restrict__ src, _Float16* __restrict__ dst,
                                     int K, int N) {
  int i = blockIdx.x * 256 + threadIdx.x;
  if (i >= K * N) return;
  int k = i / N, n = i % N;
  dst[(size_t)n * K + k] = (_Float16)src[i];
}

// transpose + pad K: dst[n*Kdst+k] = (k<Ksrc) ? src[k*N+n] : 0
__global__ void cvt_transpose_pad_kernel(const float* __restrict__ src, _Float16* __restrict__ dst,
                                         int Ksrc, int Kdst, int N) {
  int i = blockIdx.x * 256 + threadIdx.x;
  if (i >= Kdst * N) return;
  int n = i / Kdst, k = i % Kdst;
  dst[i] = (k < Ksrc) ? (_Float16)src[(size_t)k * N + n] : (_Float16)0.0f;
}

__global__ void zero_u32_kernel(uint32_t* p, int n) {
  int i = blockIdx.x * 256 + threadIdx.x;
  if (i < n) p[i] = 0u;
}

// ---------------- feature assembly ----------------
__global__ void build_X_kernel(const float* __restrict__ tf, const int* __restrict__ pos_ids,
                               const int* __restrict__ dep_ids, const int* __restrict__ e1,
                               const int* __restrict__ e2, const int* __restrict__ sdp,
                               const float* __restrict__ pos_tab, const float* __restrict__ dep_tab,
                               const float* __restrict__ dist_tab, const float* __restrict__ sdp_tab,
                               _Float16* __restrict__ X, _Float16* __restrict__ dep16) {
  int row = blockIdx.x;            // b*T + t
  int b = row >> 8, t = row & 255;
  int p = pos_ids[row], d = dep_ids[row], sp = sdp[row];
  int d1 = t - e1[b] + 10; d1 = d1 < 0 ? 0 : (d1 > 20 ? 20 : d1);
  int d2 = t - e2[b] + 10; d2 = d2 < 0 ? 0 : (d2 > 20 ? 20 : d2);
  for (int c = threadIdx.x; c < FEATP; c += 256) {
    float v;
    if      (c < 768) v = tf[(size_t)row * 768 + c];
    else if (c < 800) v = pos_tab[p * 32 + (c - 768)];
    else if (c < 832) v = dep_tab[d * 32 + (c - 800)];
    else if (c < 864) v = dist_tab[d1 * 32 + (c - 832)];
    else if (c < 896) v = dist_tab[d2 * 32 + (c - 864)];
    else if (c < 912) v = sdp_tab[sp * 16 + (c - 896)];
    else v = 0.0f;
    X[(size_t)row * FEATP + c] = (_Float16)v;
  }
  if (threadIdx.x < 32) dep16[(size_t)row * 32 + threadIdx.x] = (_Float16)dep_tab[d * 32 + threadIdx.x];
}

// ---------------- generic WMMA GEMM: out = act(A@Bt^T + bias + addC) ----------------
// A:[M,K] f16 row-major; Bt:[N,K] f16 row-major (transposed weights); K%32==0, M%16==0, N%128==0.
// B tiles staged via Tensor Data Mover (128 rows x 64B) into LDS.
__global__ void wmma_gemm_kernel(const _Float16* __restrict__ A, const _Float16* __restrict__ Bt,
                                 const float* __restrict__ bias, const float* __restrict__ addC,
                                 void* __restrict__ outp, int M, int N, int K, int act, int out_f16) {
  __shared__ __align__(16) uint32_t sAu[16 * 16];   // A chunk: 16x32 halfs, row-major
  __shared__ __align__(16) uint32_t sBu[128 * 16];  // B chunk: [128 n][32 k] halfs, row-major
  const _Float16* sA  = (const _Float16*)sAu;
  const _Float16* sBt = (const _Float16*)sBu;
  int tid = threadIdx.x, wave = tid >> 5, lane = tid & 31;
  int m0 = blockIdx.x * 16;
  int nb = blockIdx.y * 128;
  int n0 = nb + wave * 16;
  CFrag acc;
  #pragma unroll
  for (int i = 0; i < 8; ++i) acc.e[i] = 0.0f;
  const int rA = tid >> 4, cA = tid & 15;
  const int ar = lane & 15, akh = (lane >> 4) * 8;
  const int bn = lane & 15, bkh = (lane >> 4) * 8;
  const unsigned ldsB = (unsigned)(uintptr_t)(void*)sBu;  // LDS byte offset (low 32 bits)
  for (int k0 = 0; k0 < K; k0 += 32) {
    sAu[rA * 16 + cA] = *(const uint32_t*)(A + (size_t)(m0 + rA) * K + k0 + cA * 2);
    if (wave == 0) {
      tdm_load_2d(Bt + (size_t)nb * K + k0, ldsB, (unsigned)(K / 2), (unsigned)N, 128u);
      __builtin_amdgcn_s_wait_tensorcnt(0);
    }
    if (k0 + 32 < K) __builtin_prefetch(A + (size_t)m0 * K + (k0 + 32), 0, 1);
    __syncthreads();
    AFrag a, bf;
    const _Float16* arow = sA + ar * 32 + akh;
    *(uint4*)&a.e[0] = *(const uint4*)(arow);
    *(uint4*)&a.e[8] = *(const uint4*)(arow + 16);
    const _Float16* brow = sBt + (wave * 16 + bn) * 32 + bkh;
    *(uint4*)&bf.e[0] = *(const uint4*)(brow);
    *(uint4*)&bf.e[8] = *(const uint4*)(brow + 16);
    acc.v = __builtin_amdgcn_wmma_f32_16x16x32_f16(false, a.v, false, bf.v, (short)0, acc.v, false, false);
    __syncthreads();
  }
  #pragma unroll
  for (int i = 0; i < 8; ++i) {
    int mi = m0 + i + ((lane >> 4) << 3);
    int ni = n0 + (lane & 15);
    float v = acc.e[i];
    if (bias) v += bias[ni];
    if (addC) v += addC[(size_t)mi * N + ni];
    if (act == 1) v = tanhf(v);
    if (out_f16) ((_Float16*)outp)[(size_t)mi * N + ni] = (_Float16)v;
    else         ((float*)outp)[(size_t)mi * N + ni] = v;
  }
}

// ---------------- BiLSTM recurrence: one 512-thread workgroup per direction ----------------
// g[64,1024] = xw[:,t,:] + h @ Whh ; WhhT is [1024,256] f16 (pre-transposed).
__global__ void lstm_rec_kernel(const _Float16* __restrict__ xw_f, const _Float16* __restrict__ xw_b,
                                const _Float16* __restrict__ whhT_f, const _Float16* __restrict__ whhT_b,
                                _Float16* __restrict__ hs_f, _Float16* __restrict__ hs_b) {
  const int dir = blockIdx.x;
  const _Float16* xw   = dir ? xw_b   : xw_f;
  const _Float16* whhT = dir ? whhT_b : whhT_f;
  _Float16*       hs   = dir ? hs_b   : hs_f;
  __shared__ __align__(16) _Float16 hbuf[64 * 256];
  int tid = threadIdx.x, wave = tid >> 5, lane = tid & 31;
  for (int i = tid; i < 64 * 256 / 2; i += 512) ((uint32_t*)hbuf)[i] = 0u;
  __syncthreads();
  float cst[4][8], hnew[4][8];
  #pragma unroll
  for (int c = 0; c < 4; ++c)
    #pragma unroll
    for (int r = 0; r < 8; ++r) cst[c][r] = 0.0f;
  const int ar = lane & 15, akh = (lane >> 4) * 8;
  const int bkh = (lane >> 4) * 8, bn = lane & 15;
  const int cmh = (lane >> 4) * 8, cn = lane & 15;
  for (int s = 0; s < TT; ++s) {
    int t = dir ? (TT - 1 - s) : s;
    for (int c = 0; c < 4; ++c) {
      int ci = wave * 4 + c;
      int m0 = (ci >> 4) * 16, n0 = (ci & 15) * 16;
      CFrag g[4];
      #pragma unroll
      for (int q = 0; q < 4; ++q) {
        int col = q * 256 + n0 + cn;
        #pragma unroll
        for (int r = 0; r < 8; ++r)
          g[q].e[r] = (float)xw[(size_t)((m0 + r + cmh) * TT + t) * 1024 + col];
      }
      for (int kc = 0; kc < 8; ++kc) {
        AFrag a;
        const _Float16* arow = hbuf + (m0 + ar) * 256 + kc * 32 + akh;
        *(uint4*)&a.e[0] = *(const uint4*)(arow);
        *(uint4*)&a.e[8] = *(const uint4*)(arow + 16);
        #pragma unroll
        for (int q = 0; q < 4; ++q) {
          AFrag bf;
          int colb = q * 256 + n0 + bn;
          const _Float16* brow = whhT + (size_t)colb * 256 + kc * 32 + bkh;
          *(uint4*)&bf.e[0] = *(const uint4*)(brow);
          *(uint4*)&bf.e[8] = *(const uint4*)(brow + 16);
          g[q].v = __builtin_amdgcn_wmma_f32_16x16x32_f16(false, a.v, false, bf.v, (short)0, g[q].v, false, false);
        }
      }
      #pragma unroll
      for (int r = 0; r < 8; ++r) {
        float ig = sigf(g[0].e[r]);
        float fg = sigf(g[1].e[r]);
        float gg = tanhf(g[2].e[r]);
        float og = sigf(g[3].e[r]);
        float cc = fg * cst[c][r] + ig * gg;
        cst[c][r] = cc;
        hnew[c][r] = og * tanhf(cc);
      }
    }
    __syncthreads();  // all reads of old h done
    #pragma unroll
    for (int c = 0; c < 4; ++c) {
      int ci = wave * 4 + c;
      int m0 = (ci >> 4) * 16, n0 = (ci & 15) * 16;
      #pragma unroll
      for (int r = 0; r < 8; ++r) {
        int mi = m0 + r + cmh, ni = n0 + cn;
        _Float16 hv = (_Float16)hnew[c][r];
        hbuf[mi * 256 + ni] = hv;
        hs[(size_t)(mi * TT + t) * 256 + ni] = hv;
      }
    }
    __syncthreads();
  }
}

// ---------------- tree helpers ----------------
__global__ void build_xtree_kernel(const _Float16* __restrict__ hsf, const _Float16* __restrict__ hsb,
                                   const _Float16* __restrict__ dep16, _Float16* __restrict__ xt) {
  int row = blockIdx.x;
  for (int j = threadIdx.x; j < 544; j += 256) {
    _Float16 v;
    if      (j < 256) v = hsf[(size_t)row * 256 + j];
    else if (j < 512) v = hsb[(size_t)row * 256 + (j - 256)];
    else              v = dep16[(size_t)row * 32 + (j - 512)];
    xt[(size_t)row * 544 + j] = v;
  }
}

__global__ void gather_par_kernel(const float* __restrict__ xwf, const int* __restrict__ heads,
                                  float* __restrict__ out) {
  int row = blockIdx.x, k = threadIdx.x;
  int b = row >> 8, h = heads[row];
  out[(size_t)row * 256 + k] = xwf[(size_t)(b * 256 + h) * 256 + k];
}

__global__ void tree_f_kernel(const float* __restrict__ hUf, const float* __restrict__ xwfpar,
                              const _Float16* __restrict__ ht, const float* __restrict__ ct,
                              const int* __restrict__ heads, float* __restrict__ hsum,
                              float* __restrict__ fc) {
  int row = blockIdx.x, k = threadIdx.x;
  int b = row >> 8, t = row & 255;
  int hd = heads[row];
  size_t i = (size_t)row * 256 + k;
  float f = sigf(xwfpar[i] + hUf[i]);
  if (hd != t) {  // mask drops root self-loop
    size_t pi = (size_t)(b * 256 + hd) * 256 + k;
    atomicAdd(&hsum[pi], (float)ht[i]);
    atomicAdd(&fc[pi], f * ct[i]);
  }
}

__global__ void tree_update_kernel(const float* __restrict__ iou, const float* __restrict__ fc,
                                   float* __restrict__ ct, _Float16* __restrict__ ht) {
  int row = blockIdx.x, k = threadIdx.x;
  size_t base = (size_t)row * 768;
  float iv = iou[base + k], ov = iou[base + 256 + k], uv = iou[base + 512 + k];
  float c2 = sigf(iv) * tanhf(uv) + fc[(size_t)row * 256 + k];
  float h2 = sigf(ov) * tanhf(c2);
  ct[(size_t)row * 256 + k] = c2;
  ht[(size_t)row * 256 + k] = (_Float16)h2;
}

// ---------------- classifier ----------------
__global__ void classifier_kernel(const _Float16* __restrict__ ht, const int* __restrict__ root_idx,
                                  const float* __restrict__ W1, const float* __restrict__ b1,
                                  const float* __restrict__ W2, const float* __restrict__ b2,
                                  float* __restrict__ out) {
  __shared__ float hid[256];
  __shared__ float rh[256];
  int b = blockIdx.x, j = threadIdx.x;
  int row = b * 256 + root_idx[b];
  rh[j] = (float)ht[(size_t)row * 256 + j];
  __syncthreads();
  float s = b1[j];
  for (int k = 0; k < 256; ++k) s += rh[k] * W1[k * 256 + j];
  hid[j] = s > 0.0f ? s : 0.0f;
  __syncthreads();
  if (j < 10) {
    float l = b2[j];
    for (int kk = 0; kk < 256; ++kk) l += hid[kk] * W2[kk * 10 + j];
    out[b * 10 + j] = l;
  }
}

// ---------------- host launcher ----------------
extern "C" void kernel_launch(void* const* d_in, const int* in_sizes, int n_in,
                              void* d_out, int out_size, void* d_ws, size_t ws_size,
                              hipStream_t stream) {
  const float* token_features = (const float*)d_in[0];
  const int*   pos_ids   = (const int*)d_in[1];
  const int*   dep_ids   = (const int*)d_in[2];
  const int*   e1_idx    = (const int*)d_in[3];
  const int*   e2_idx    = (const int*)d_in[4];
  const int*   sdp_pos   = (const int*)d_in[5];
  const int*   dep_heads = (const int*)d_in[6];
  const int*   root_idx  = (const int*)d_in[7];
  const float* pos_table = (const float*)d_in[8];
  const float* dep_table = (const float*)d_in[9];
  const float* dist_table= (const float*)d_in[10];
  const float* sdp_table = (const float*)d_in[11];
  const float* W_ft = (const float*)d_in[12];
  const float* b_ft = (const float*)d_in[13];
  const float* Wih_f = (const float*)d_in[14];
  const float* Whh_f = (const float*)d_in[15];
  const float* bl_f  = (const float*)d_in[16];
  const float* Wih_b = (const float*)d_in[17];
  const float* Whh_b = (const float*)d_in[18];
  const float* bl_b  = (const float*)d_in[19];
  const float* W_iou = (const float*)d_in[20];
  const float* U_iou = (const float*)d_in[21];
  const float* b_iou = (const float*)d_in[22];
  const float* Wf_t  = (const float*)d_in[23];
  const float* Uf_t  = (const float*)d_in[24];
  const float* bf_t  = (const float*)d_in[25];
  const float* W1 = (const float*)d_in[26];
  const float* b1 = (const float*)d_in[27];
  const float* W2 = (const float*)d_in[28];
  const float* b2 = (const float*)d_in[29];

  char* ws = (char*)d_ws;
  size_t off = 0;
  auto alloc = [&](size_t bytes) { size_t o = off; off += (bytes + 255) & ~(size_t)255; return o; };
  const size_t R = (size_t)BB * TT;  // 16384 rows

  // persistent transposed f16 weights + dep embedding
  size_t o_wft  = alloc((size_t)FEATP * FTD * 2);     // [512][928]
  size_t o_wihf = alloc((size_t)512 * 1024 * 2);      // [1024][512]
  size_t o_wihb = alloc((size_t)512 * 1024 * 2);
  size_t o_whhf = alloc((size_t)256 * 1024 * 2);      // [1024][256]
  size_t o_whhb = alloc((size_t)256 * 1024 * 2);
  size_t o_wiou = alloc((size_t)544 * 768 * 2);       // [768][544]
  size_t o_wf   = alloc((size_t)544 * 256 * 2);       // [256][544]
  size_t o_uiou = alloc((size_t)256 * 768 * 2);       // [768][256]
  size_t o_uf   = alloc((size_t)256 * 256 * 2);       // [256][256]
  size_t o_dep16= alloc(R * 32 * 2);
  // phase-1 arena (dead after recurrence; re-used for xWiou / iou)
  size_t p1    = off;
  size_t o_X   = alloc(R * FEATP * 2);
  size_t o_T   = alloc(R * FTD * 2);
  size_t o_xwf = alloc(R * 1024 * 2);
  size_t o_xwb = alloc(R * 1024 * 2);
  // phase-2 buffers
  size_t o_hsf   = alloc(R * 256 * 2);
  size_t o_hsb   = alloc(R * 256 * 2);
  size_t o_xtree = alloc(R * 544 * 2);
  size_t o_xwfpar= alloc(R * 256 * 4);
  size_t o_ht16  = alloc(R * 256 * 2);
  size_t o_ct    = alloc(R * 256 * 4);
  size_t o_hsum  = alloc(R * 256 * 4);
  size_t o_hsum16= alloc(R * 256 * 2);
  size_t o_fc    = alloc(R * 256 * 4);
  size_t o_hUf   = alloc(R * 256 * 4);
  size_t o_xwftmp= alloc(R * 256 * 4);
  // aliases into dead phase-1 arena
  size_t o_xwiou = p1;                       // R*768*4 = 48MB
  size_t o_iou   = p1 + R * 768 * 4;         // another 48MB (fits in 114MB arena)
  (void)ws_size; (void)in_sizes; (void)n_in; (void)out_size;

  auto F16 = [&](size_t o) { return (_Float16*)(ws + o); };
  auto F32 = [&](size_t o) { return (float*)(ws + o); };

  // weight transpose + convert
  auto cvtT = [&](const float* src, size_t dsto, int K, int N) {
    cvt_transpose_kernel<<<(K * N + 255) / 256, 256, 0, stream>>>(src, F16(dsto), K, N);
  };
  cvt_transpose_pad_kernel<<<(FEATP * FTD + 255) / 256, 256, 0, stream>>>(W_ft, F16(o_wft), 912, FEATP, FTD);
  cvtT(Wih_f, o_wihf, 512, 1024);
  cvtT(Wih_b, o_wihb, 512, 1024);
  cvtT(Whh_f, o_whhf, 256, 1024);
  cvtT(Whh_b, o_whhb, 256, 1024);
  cvtT(W_iou, o_wiou, 544, 768);
  cvtT(Wf_t,  o_wf,   544, 256);
  cvtT(U_iou, o_uiou, 256, 768);
  cvtT(Uf_t,  o_uf,   256, 256);

  // features + transform
  build_X_kernel<<<(int)R, 256, 0, stream>>>(token_features, pos_ids, dep_ids, e1_idx, e2_idx,
                                             sdp_pos, pos_table, dep_table, dist_table, sdp_table,
                                             F16(o_X), F16(o_dep16));
  wmma_gemm_kernel<<<dim3(R / 16, FTD / 128), 256, 0, stream>>>(
      F16(o_X), F16(o_wft), b_ft, nullptr, (void*)F16(o_T), (int)R, FTD, FEATP, /*tanh*/1, /*f16*/1);

  // LSTM input projections
  wmma_gemm_kernel<<<dim3(R / 16, 1024 / 128), 256, 0, stream>>>(
      F16(o_T), F16(o_wihf), bl_f, nullptr, (void*)F16(o_xwf), (int)R, 1024, 512, 0, 1);
  wmma_gemm_kernel<<<dim3(R / 16, 1024 / 128), 256, 0, stream>>>(
      F16(o_T), F16(o_wihb), bl_b, nullptr, (void*)F16(o_xwb), (int)R, 1024, 512, 0, 1);

  // recurrence, both directions concurrently
  lstm_rec_kernel<<<2, 512, 0, stream>>>(F16(o_xwf), F16(o_xwb), F16(o_whhf), F16(o_whhb),
                                         F16(o_hsf), F16(o_hsb));

  // tree inputs
  build_xtree_kernel<<<(int)R, 256, 0, stream>>>(F16(o_hsf), F16(o_hsb), F16(o_dep16), F16(o_xtree));
  wmma_gemm_kernel<<<dim3(R / 16, 768 / 128), 256, 0, stream>>>(
      F16(o_xtree), F16(o_wiou), b_iou, nullptr, (void*)F32(o_xwiou), (int)R, 768, 544, 0, 0);
  wmma_gemm_kernel<<<dim3(R / 16, 256 / 128), 256, 0, stream>>>(
      F16(o_xtree), F16(o_wf), bf_t, nullptr, (void*)F32(o_xwftmp), (int)R, 256, 544, 0, 0);
  gather_par_kernel<<<(int)R, 256, 0, stream>>>(F32(o_xwftmp), dep_heads, F32(o_xwfpar));

  // zero tree state
  int nS = (int)(R * 256);                       // 4.19M f32 elements
  zero_u32_kernel<<<(nS / 2 + 255) / 256, 256, 0, stream>>>((uint32_t*)F16(o_ht16), nS / 2);
  zero_u32_kernel<<<(nS + 255) / 256, 256, 0, stream>>>((uint32_t*)F32(o_ct), nS);

  // 9 tree sweeps
  for (int it = 0; it < 9; ++it) {
    zero_u32_kernel<<<(nS + 255) / 256, 256, 0, stream>>>((uint32_t*)F32(o_hsum), nS);
    zero_u32_kernel<<<(nS + 255) / 256, 256, 0, stream>>>((uint32_t*)F32(o_fc), nS);
    wmma_gemm_kernel<<<dim3(R / 16, 256 / 128), 256, 0, stream>>>(
        F16(o_ht16), F16(o_uf), nullptr, nullptr, (void*)F32(o_hUf), (int)R, 256, 256, 0, 0);
    tree_f_kernel<<<(int)R, 256, 0, stream>>>(F32(o_hUf), F32(o_xwfpar), F16(o_ht16), F32(o_ct),
                                              dep_heads, F32(o_hsum), F32(o_fc));
    cvt_f32_f16_kernel<<<(nS + 255) / 256, 256, 0, stream>>>(F32(o_hsum), F16(o_hsum16), nS);
    wmma_gemm_kernel<<<dim3(R / 16, 768 / 128), 256, 0, stream>>>(
        F16(o_hsum16), F16(o_uiou), nullptr, F32(o_xwiou), (void*)F32(o_iou), (int)R, 768, 256, 0, 0);
    tree_update_kernel<<<(int)R, 256, 0, stream>>>(F32(o_iou), F32(o_fc), F32(o_ct), F16(o_ht16));
  }

  // classifier
  classifier_kernel<<<BB, 256, 0, stream>>>(F16(o_ht16), root_idx, W1, b1, W2, b2, (float*)d_out);
}
